// Network_2491081031828
// MI455X (gfx1250) — compile-verified, matrix operand order checked by reading, not run on hardware
//
#include <hip/hip_runtime.h>
#include <hip/hip_bf16.h>
#include <math.h>

// ---------------------------------------------------------------------------
// Network: 3x (conv5x5 + relu + maxpool2) -> concat(histc) -> 4-layer MLP.
// Matrix math on v_wmma_f32_16x16x32_f16; conv2 input staged to LDS with
// global_load_async_to_lds_b128 (ASYNCcnt); im2col K->LDS-offset via a u16
// table so the inner loop has no divisions; dead K slots read zeroed LDS rows.
// ---------------------------------------------------------------------------

typedef __attribute__((ext_vector_type(16))) _Float16 v16h;
typedef __attribute__((ext_vector_type(8)))  float    v8f;

union HFrag { v16h v; _Float16 h[16]; };
union FFrag { v8f  v; float    f[8];  };

#define NBINS    100
#define FEAT_PAD 832   // 726 conv feats + 100 hist + 6 zero pad (== 26*32)

// K index held by (lane, half j) of a 16-bit 16x32 WMMA A/B fragment:
//   k = 16*(j>=8) + 8*(lane>=16) + (j&7)
__device__ __forceinline__ v8f wmma16(v16h a, v16h b, v8f c) {
    return __builtin_amdgcn_wmma_f32_16x16x32_f16(
        /*neg_a=*/false, a, /*neg_b=*/false, b,
        /*c_mod=*/(short)0, c, /*reuse_a=*/false, /*reuse_b=*/false);
}

// ---------------------------------------------------------------------------
// Histogram: one block per sample, LDS-atomic 100 bins of t*255.
// Writes feat[s][726..825] and zeros pad columns 826..831.
// ---------------------------------------------------------------------------
__global__ __launch_bounds__(256) void hist_kernel(const float* __restrict__ t,
                                                   float* __restrict__ feat) {
    __shared__ unsigned int bins[NBINS];
    const int s   = blockIdx.x;
    const int tid = threadIdx.x;
    for (int i = tid; i < NBINS; i += 256) bins[i] = 0u;
    __syncthreads();
    const float* ts = t + (size_t)s * (116 * 116);
    for (int i = tid; i < 116 * 116; i += 256) {
        float v = ts[i] * 255.0f;
        if (v >= 0.0f && v <= 255.0f) {
            int b = (int)floorf(v * (100.0f / 255.0f));
            b = b < 0 ? 0 : (b > NBINS - 1 ? NBINS - 1 : b);
            atomicAdd(&bins[b], 1u);
        }
    }
    __syncthreads();
    float* fs = feat + (size_t)s * FEAT_PAD;
    for (int i = tid; i < 106; i += 256)
        fs[726 + i] = (i < NBINS) ? (float)bins[i] : 0.0f;
}

// ---------------------------------------------------------------------------
// Conv1 (1->12, 5x5, 116->112) + ReLU + pool -> p1[12,56,56] f16.
// 8 waves/block, waves 0..6 = 7 x-strips of one pooled row. K=25 pad 32.
// smem rows 0..5 = input rows y0..y0+5 (width 116), rows 6..8 zeroed
// (dead-K target). tab[k] = LDS byte offset of tap k.
// ---------------------------------------------------------------------------
__global__ __launch_bounds__(256) void conv1_kernel(const float* __restrict__ t,
                                                    const float* __restrict__ w1,
                                                    const float* __restrict__ b1,
                                                    _Float16* __restrict__ p1) {
    __shared__ _Float16 smem[9 * 116];
    __shared__ unsigned short tab[32];
    const int pr  = blockIdx.x;        // pooled row 0..55
    const int s   = blockIdx.y;
    const int tid = threadIdx.x;
    const int y0  = 2 * pr;
    const float* ts = t + (size_t)s * (116 * 116);
    for (int i = tid; i < 6 * 116; i += 256) {
        int yy = i / 116, x = i - yy * 116;
        smem[i] = (_Float16)ts[(y0 + yy) * 116 + x];
    }
    for (int i = tid; i < 3 * 116; i += 256) smem[6 * 116 + i] = (_Float16)0.0f;
    if (tid < 32) {
        int off;
        if (tid < 25) { int dy = tid / 5, dx = tid - dy * 5; off = (dy * 116 + dx) * 2; }
        else off = 6 * 116 * 2;
        tab[tid] = (unsigned short)off;
    }
    // gfx1250 prefetch of the next pooled row's fresh input rows (L2 warm-up).
    if (tid < 232) {
        int row = y0 + 6 + (tid / 116);
        if (row < 116) __builtin_prefetch(ts + row * 116 + (tid % 116), 0, 1);
    }
    __syncthreads();

    const int wave = tid >> 5;
    const int lane = tid & 31;
    if (wave >= 7) return;             // wave-uniform
    const int x0  = wave * 16;
    const int l16 = lane & 15;
    const int hi  = lane >> 4;
    const int kh  = hi << 3;

    const char*  apb = (const char*)(smem + x0 + l16);
    const float* wp  = w1 + (l16 < 12 ? l16 : 0) * 25;

    HFrag a0, a1, bf;
    #pragma unroll
    for (int j = 0; j < 16; ++j) {
        int k   = ((j & 8) << 1) + (j & 7) + kh;
        int off = tab[k];
        a0.h[j] = *(const _Float16*)(apb + off);
        a1.h[j] = *(const _Float16*)(apb + off + 232);   // +1 input row
        bf.h[j] = (k < 25) ? (_Float16)wp[k] : (_Float16)0.0f;
    }
    FFrag d0, d1;
    #pragma unroll
    for (int r = 0; r < 8; ++r) { d0.f[r] = 0.0f; d1.f[r] = 0.0f; }
    d0.v = wmma16(a0.v, bf.v, d0.v);
    d1.v = wmma16(a1.v, bf.v, d1.v);

    if (l16 < 12) {
        float bias = b1[l16];
        _Float16* pp = p1 + (((size_t)s * 12 + l16) * 56 + pr) * 56;
        #pragma unroll
        for (int i = 0; i < 4; ++i) {
            float v = fmaxf(fmaxf(d0.f[2 * i], d0.f[2 * i + 1]),
                            fmaxf(d1.f[2 * i], d1.f[2 * i + 1]));
            pp[(x0 >> 1) + (hi << 2) + i] = (_Float16)fmaxf(v + bias, 0.0f);
        }
    }
}

// ---------------------------------------------------------------------------
// Conv2 (12->12, 5x5, 56->52) + ReLU + pool -> p2[12,26,26] f16.
// 4 waves = 4 x-strips. K = 300 -> 10 k-steps (last peeled for weight guard).
// Input tile staged via global_load_async_to_lds_b128 into width-72 padded
// rows; cols 56..71 and rows 72..74 zeroed (x-overrun + dead-K targets).
// ---------------------------------------------------------------------------
__global__ __launch_bounds__(128) void conv2_kernel(const _Float16* __restrict__ p1,
                                                    const float* __restrict__ w2,
                                                    const float* __restrict__ b2,
                                                    _Float16* __restrict__ p2) {
    __shared__ _Float16 smem[75 * 72];       // [c*6+yy][72]
    __shared__ unsigned short tab[320];
    const int pr  = blockIdx.x;              // pooled row 0..25
    const int s   = blockIdx.y;
    const int tid = threadIdx.x;
    const int y0  = 2 * pr;

    // Async DMA: 72 rows x 112B (= 7 x b128 each), all 16B aligned.
    {
        const unsigned long long src0 =
            (unsigned long long)((const char*)p1 + (size_t)s * (12 * 56 * 56 * 2)
                                                 + (size_t)y0 * 112);
        const unsigned lds0 = (unsigned)(size_t)&smem[0];
        for (int i = tid; i < 504; i += 128) {
            int row = i / 7, seg = i - row * 7;
            int c = row / 6, yy = row - c * 6;
            unsigned ldsa = lds0 + (unsigned)(row * 144 + seg * 16);
            unsigned long long ga = src0 + (unsigned)(c * 6272 + yy * 112 + seg * 16);
            asm volatile("global_load_async_to_lds_b128 %0, %1, off"
                         :: "v"(ldsa), "v"(ga) : "memory");
        }
        for (int i = tid; i < 72 * 16; i += 128) {          // cols 56..71
            int row = i >> 4, col = 56 + (i & 15);
            smem[row * 72 + col] = (_Float16)0.0f;
        }
        for (int i = tid; i < 3 * 72; i += 128)             // rows 72..74
            smem[72 * 72 + i] = (_Float16)0.0f;
        for (int i = tid; i < 320; i += 128) {              // K -> LDS offset
            int off;
            if (i < 300) {
                int c = i / 25, rr = i - c * 25, dy = rr / 5, dx = rr - dy * 5;
                off = ((c * 6 + dy) * 72 + dx) * 2;
            } else off = 72 * 72 * 2;
            tab[i] = (unsigned short)off;
        }
        asm volatile("s_wait_asynccnt 0x0" ::: "memory");
    }
    __syncthreads();

    const int wave = tid >> 5;               // 0..3, all active
    const int lane = tid & 31;
    const int x0  = wave * 16;
    const int l16 = lane & 15;
    const int hi  = lane >> 4;
    const int kh  = hi << 3;

    const char*  apb = (const char*)(smem + x0 + l16);
    const float* wp  = w2 + (l16 < 12 ? l16 : 0) * 300;

    FFrag d0, d1;
    #pragma unroll
    for (int r = 0; r < 8; ++r) { d0.f[r] = 0.0f; d1.f[r] = 0.0f; }

    for (int kk = 0; kk < 9; ++kk) {         // k <= 287 < 300: no weight guard
        HFrag af0, af1, bf;
        #pragma unroll
        for (int j = 0; j < 16; ++j) {
            int k   = kk * 32 + ((j & 8) << 1) + (j & 7) + kh;
            int off = tab[k];
            af0.h[j] = *(const _Float16*)(apb + off);
            af1.h[j] = *(const _Float16*)(apb + off + 144);
            bf.h[j]  = (_Float16)wp[k];
        }
        d0.v = wmma16(af0.v, bf.v, d0.v);
        d1.v = wmma16(af1.v, bf.v, d1.v);
    }
    {                                        // peeled kk = 9
        HFrag af0, af1, bf;
        #pragma unroll
        for (int j = 0; j < 16; ++j) {
            int k   = 288 + ((j & 8) << 1) + (j & 7) + kh;
            int off = tab[k];
            af0.h[j] = *(const _Float16*)(apb + off);
            af1.h[j] = *(const _Float16*)(apb + off + 144);
            bf.h[j]  = (k < 300) ? (_Float16)wp[k] : (_Float16)0.0f;
        }
        d0.v = wmma16(af0.v, bf.v, d0.v);
        d1.v = wmma16(af1.v, bf.v, d1.v);
    }

    if (l16 < 12) {
        float bias = b2[l16];
        _Float16* pp = p2 + (((size_t)s * 12 + l16) * 26 + pr) * 26;
        #pragma unroll
        for (int i = 0; i < 4; ++i) {
            int px = (x0 >> 1) + (hi << 2) + i;
            if (px < 26) {
                float v = fmaxf(fmaxf(d0.f[2 * i], d0.f[2 * i + 1]),
                                fmaxf(d1.f[2 * i], d1.f[2 * i + 1]));
                pp[px] = (_Float16)fmaxf(v + bias, 0.0f);
            }
        }
    }
}

// ---------------------------------------------------------------------------
// Conv3 (12->6, 5x5, 26->22) + ReLU + pool -> feat[s][0..725] f32 flatten.
// 2 waves = 2 x-strips. Width padded 26->40, rows 72..74 zeroed.
// ---------------------------------------------------------------------------
__global__ __launch_bounds__(64) void conv3_kernel(const _Float16* __restrict__ p2,
                                                   const float* __restrict__ w3,
                                                   const float* __restrict__ b3,
                                                   float* __restrict__ feat) {
    __shared__ _Float16 smem[75 * 40];
    __shared__ unsigned short tab[320];
    const int pr  = blockIdx.x;              // pooled row 0..10
    const int s   = blockIdx.y;
    const int tid = threadIdx.x;
    const int y0  = 2 * pr;
    const _Float16* ps = p2 + (size_t)s * (12 * 26 * 26);
    for (int i = tid; i < 12 * 6 * 40; i += 64) {
        int c = i / 240, rem = i - c * 240;
        int yy = rem / 40, x = rem - yy * 40;
        smem[i] = (x < 26) ? ps[(c * 26 + y0 + yy) * 26 + x] : (_Float16)0.0f;
    }
    for (int i = tid; i < 3 * 40; i += 64) smem[72 * 40 + i] = (_Float16)0.0f;
    for (int i = tid; i < 320; i += 64) {
        int off;
        if (i < 300) {
            int c = i / 25, rr = i - c * 25, dy = rr / 5, dx = rr - dy * 5;
            off = ((c * 6 + dy) * 40 + dx) * 2;
        } else off = 72 * 40 * 2;
        tab[i] = (unsigned short)off;
    }
    __syncthreads();

    const int wave = tid >> 5;               // 0..1
    const int lane = tid & 31;
    const int x0  = wave * 16;
    const int l16 = lane & 15;
    const int hi  = lane >> 4;
    const int kh  = hi << 3;

    const char*  apb = (const char*)(smem + x0 + l16);
    const float* wp  = w3 + (l16 < 6 ? l16 : 0) * 300;

    FFrag d0, d1;
    #pragma unroll
    for (int r = 0; r < 8; ++r) { d0.f[r] = 0.0f; d1.f[r] = 0.0f; }

    for (int kk = 0; kk < 9; ++kk) {
        HFrag af0, af1, bf;
        #pragma unroll
        for (int j = 0; j < 16; ++j) {
            int k   = kk * 32 + ((j & 8) << 1) + (j & 7) + kh;
            int off = tab[k];
            af0.h[j] = *(const _Float16*)(apb + off);
            af1.h[j] = *(const _Float16*)(apb + off + 80);
            bf.h[j]  = (_Float16)wp[k];
        }
        d0.v = wmma16(af0.v, bf.v, d0.v);
        d1.v = wmma16(af1.v, bf.v, d1.v);
    }
    {
        HFrag af0, af1, bf;
        #pragma unroll
        for (int j = 0; j < 16; ++j) {
            int k   = 288 + ((j & 8) << 1) + (j & 7) + kh;
            int off = tab[k];
            af0.h[j] = *(const _Float16*)(apb + off);
            af1.h[j] = *(const _Float16*)(apb + off + 80);
            bf.h[j]  = (k < 300) ? (_Float16)wp[k] : (_Float16)0.0f;
        }
        d0.v = wmma16(af0.v, bf.v, d0.v);
        d1.v = wmma16(af1.v, bf.v, d1.v);
    }

    if (l16 < 6) {
        float bias = b3[l16];
        float* fs = feat + (size_t)s * FEAT_PAD + l16 * 121 + pr * 11;
        #pragma unroll
        for (int i = 0; i < 4; ++i) {
            int px = (x0 >> 1) + (hi << 2) + i;
            if (px < 11) {
                float v = fmaxf(fmaxf(d0.f[2 * i], d0.f[2 * i + 1]),
                                fmaxf(d1.f[2 * i], d1.f[2 * i + 1]));
                fs[px] = fmaxf(v + bias, 0.0f);
            }
        }
    }
}

// ---------------------------------------------------------------------------
// FC1: [1024,832(pad)] x [826->112]^T, ReLU -> a1 f16. 64 blocks x 8 waves
// (7 N-tiles active), 26 k-steps; feat pad columns are zero.
// ---------------------------------------------------------------------------
__global__ __launch_bounds__(256) void fc1_kernel(const float* __restrict__ feat,
                                                  const float* __restrict__ fw1,
                                                  const float* __restrict__ fb1,
                                                  _Float16* __restrict__ a1) {
    const int m0   = blockIdx.x * 16;
    const int wave = threadIdx.x >> 5;
    const int lane = threadIdx.x & 31;
    if (wave >= 7) return;
    const int n0  = wave * 16;
    const int l16 = lane & 15;
    const int hi  = lane >> 4;
    const int n   = n0 + l16;

    FFrag d;
    #pragma unroll
    for (int r = 0; r < 8; ++r) d.f[r] = 0.0f;

    for (int kk = 0; kk < 26; ++kk) {
        HFrag af, bf;
        #pragma unroll
        for (int j = 0; j < 16; ++j) {
            int k = kk * 32 + ((j & 8) << 1) + (hi << 3) + (j & 7);
            af.h[j] = (_Float16)feat[(size_t)(m0 + l16) * FEAT_PAD + k];
            bf.h[j] = (n < 112 && k < 826) ? (_Float16)fw1[(size_t)n * 826 + k]
                                           : (_Float16)0.0f;
        }
        d.v = wmma16(af.v, bf.v, d.v);
    }
    if (n < 112) {
        float bias = fb1[n];
        #pragma unroll
        for (int r = 0; r < 8; ++r) {
            int m = m0 + r + (hi << 3);
            a1[(size_t)m * 112 + n] = (_Float16)fmaxf(d.f[r] + bias, 0.0f);
        }
    }
}

// ---------------------------------------------------------------------------
// FC2: [1024,112] x [112->56]^T, ReLU -> a2 f16.
// ---------------------------------------------------------------------------
__global__ __launch_bounds__(128) void fc2_kernel(const _Float16* __restrict__ a1,
                                                  const float* __restrict__ fw2,
                                                  const float* __restrict__ fb2,
                                                  _Float16* __restrict__ a2) {
    const int m0   = blockIdx.x * 16;
    const int wave = threadIdx.x >> 5;
    const int lane = threadIdx.x & 31;
    const int n0  = wave * 16;
    const int l16 = lane & 15;
    const int hi  = lane >> 4;
    const int n   = n0 + l16;

    FFrag d;
    #pragma unroll
    for (int r = 0; r < 8; ++r) d.f[r] = 0.0f;

    for (int kk = 0; kk < 4; ++kk) {
        HFrag af, bf;
        #pragma unroll
        for (int j = 0; j < 16; ++j) {
            int k = kk * 32 + ((j & 8) << 1) + (hi << 3) + (j & 7);
            af.h[j] = (k < 112) ? a1[(size_t)(m0 + l16) * 112 + k] : (_Float16)0.0f;
            bf.h[j] = (n < 56 && k < 112) ? (_Float16)fw2[(size_t)n * 112 + k]
                                          : (_Float16)0.0f;
        }
        d.v = wmma16(af.v, bf.v, d.v);
    }
    if (n < 56) {
        float bias = fb2[n];
        #pragma unroll
        for (int r = 0; r < 8; ++r) {
            int m = m0 + r + (hi << 3);
            a2[(size_t)m * 56 + n] = (_Float16)fmaxf(d.f[r] + bias, 0.0f);
        }
    }
}

// ---------------------------------------------------------------------------
// FC3: [1024,56] x [56->12]^T, ReLU -> a3 f16. One wave per block.
// ---------------------------------------------------------------------------
__global__ __launch_bounds__(32) void fc3_kernel(const _Float16* __restrict__ a2,
                                                 const float* __restrict__ fw3,
                                                 const float* __restrict__ fb3,
                                                 _Float16* __restrict__ a3) {
    const int m0   = blockIdx.x * 16;
    const int lane = threadIdx.x & 31;
    const int l16 = lane & 15;
    const int hi  = lane >> 4;
    const int n   = l16;

    FFrag d;
    #pragma unroll
    for (int r = 0; r < 8; ++r) d.f[r] = 0.0f;

    #pragma unroll
    for (int kk = 0; kk < 2; ++kk) {
        HFrag af, bf;
        #pragma unroll
        for (int j = 0; j < 16; ++j) {
            int k = kk * 32 + ((j & 8) << 1) + (hi << 3) + (j & 7);
            af.h[j] = (k < 56) ? a2[(size_t)(m0 + l16) * 56 + k] : (_Float16)0.0f;
            bf.h[j] = (n < 12 && k < 56) ? (_Float16)fw3[(size_t)n * 56 + k]
                                         : (_Float16)0.0f;
        }
        d.v = wmma16(af.v, bf.v, d.v);
    }
    if (n < 12) {
        float bias = fb3[n];
        #pragma unroll
        for (int r = 0; r < 8; ++r) {
            int m = m0 + r + (hi << 3);
            a3[(size_t)m * 12 + n] = (_Float16)fmaxf(d.f[r] + bias, 0.0f);
        }
    }
}

// ---------------------------------------------------------------------------
// Output head: [1024,12] x [12->2]^T + bias (tiny; scalar f32).
// ---------------------------------------------------------------------------
__global__ __launch_bounds__(256) void out_kernel(const _Float16* __restrict__ a3,
                                                  const float* __restrict__ ow,
                                                  const float* __restrict__ ob,
                                                  float* __restrict__ out) {
    int idx = blockIdx.x * 256 + threadIdx.x;
    if (idx >= 1024 * 2) return;
    int s = idx >> 1, o = idx & 1;
    float acc = ob[o];
    #pragma unroll
    for (int k = 0; k < 12; ++k)
        acc += (float)a3[s * 12 + k] * ow[o * 12 + k];
    out[idx] = acc;
}

// ---------------------------------------------------------------------------
extern "C" void kernel_launch(void* const* d_in, const int* in_sizes, int n_in,
                              void* d_out, int out_size, void* d_ws, size_t ws_size,
                              hipStream_t stream) {
    (void)in_sizes; (void)n_in; (void)out_size; (void)ws_size;
    const float* t   = (const float*)d_in[0];
    const float* w1  = (const float*)d_in[1];
    const float* b1  = (const float*)d_in[2];
    const float* w2  = (const float*)d_in[3];
    const float* b2  = (const float*)d_in[4];
    const float* w3  = (const float*)d_in[5];
    const float* b3  = (const float*)d_in[6];
    const float* fw1 = (const float*)d_in[7];
    const float* fb1 = (const float*)d_in[8];
    const float* fw2 = (const float*)d_in[9];
    const float* fb2 = (const float*)d_in[10];
    const float* fw3 = (const float*)d_in[11];
    const float* fb3 = (const float*)d_in[12];
    const float* ow  = (const float*)d_in[13];
    const float* ob  = (const float*)d_in[14];

    char* ws = (char*)d_ws;
    size_t off = 0;
    auto carve = [&](size_t bytes) {
        void* p = ws + off;
        off = (off + bytes + 255) & ~(size_t)255;
        return p;
    };
    float*    feat = (float*)   carve((size_t)1024 * FEAT_PAD * 4);       // 3.4 MB
    _Float16* p1   = (_Float16*)carve((size_t)1024 * 12 * 56 * 56 * 2);   // 77 MB
    _Float16* p2   = (_Float16*)carve((size_t)1024 * 12 * 26 * 26 * 2);   // 17 MB
    _Float16* a1   = (_Float16*)carve((size_t)1024 * 112 * 2);
    _Float16* a2   = (_Float16*)carve((size_t)1024 * 56 * 2);
    _Float16* a3   = (_Float16*)carve((size_t)1024 * 12 * 2);

    hist_kernel <<<1024,            256, 0, stream>>>(t, feat);
    conv1_kernel<<<dim3(56, 1024),  256, 0, stream>>>(t,  w1, b1, p1);
    conv2_kernel<<<dim3(26, 1024),  128, 0, stream>>>(p1, w2, b2, p2);
    conv3_kernel<<<dim3(11, 1024),   64, 0, stream>>>(p2, w3, b3, feat);
    fc1_kernel  <<<64,              256, 0, stream>>>(feat, fw1, fb1, a1);
    fc2_kernel  <<<64,              128, 0, stream>>>(a1,   fw2, fb2, a2);
    fc3_kernel  <<<64,               32, 0, stream>>>(a2,   fw3, fb3, a3);
    out_kernel  <<<8,               256, 0, stream>>>(a3, ow, ob, (float*)d_out);
}
